// TransformerHNHNLayer_2576980378146
// MI455X (gfx1250) — matrix operand order repaired and checked by reading, not compiled, not run on hardware
//
#include <hip/hip_runtime.h>
#include <cstdint>
#include <cstddef>

#define E_N   20000
#define K_N   16
#define NV_N  80000
#define DW_N  48

typedef __attribute__((ext_vector_type(16))) __bf16 v16bf;
typedef __attribute__((ext_vector_type(8)))  float  v8f;

struct alignas(16) B16v { unsigned u[4]; };
struct B32v { B16v lo, hi; };

__device__ inline unsigned short f2bf(float f){
  unsigned u = __float_as_uint(f);
  u += 0x7FFFu + ((u >> 16) & 1u);   // round-to-nearest-even
  return (unsigned short)(u >> 16);
}

// A-fragment (16x32 bf16, MxK): lane row = lane&15; half=lane>>4:
//   K in [kb+8h, kb+8h+8) and [kb+16+8h, kb+16+8h+8)
__device__ inline v16bf ldfragA(const unsigned short* row, int kb){
  int half = (threadIdx.x >> 4) & 1;
  const unsigned short* p = row + kb + half * 8;
  B32v f;
  f.lo = *(const B16v*)(p);
  f.hi = *(const B16v*)(p + 16);
  return __builtin_bit_cast(v16bf, f);
}

// B-fragment (32x16 bf16, KxN) from transposed storage WT[n][k]:
// lane col = lane&15; half=lane>>4: K in [kb+8h, kb+8h+8) and [kb+16+8h, kb+16+8h+8)
__device__ inline v16bf ldfragB(const unsigned short* row, int kb){
  int half = (threadIdx.x >> 4) & 1;
  const unsigned short* p = row + kb + half * 8;
  B32v f;
  f.lo = *(const B16v*)(p);
  f.hi = *(const B16v*)(p + 16);
  return __builtin_bit_cast(v16bf, f);
}

// One 16x16 output tile per wave; 8 waves cover N=128. K = 32*ktiles.
__device__ inline v8f mm16(const unsigned short* sA, int lda,
                           const unsigned short* BT, int ldb, int ktiles){
  int lane = threadIdx.x & 31, wv = threadIdx.x >> 5;
  const unsigned short* arow = sA + (lane & 15) * lda;
  const unsigned short* brow = BT + (wv * 16 + (lane & 15)) * ldb;
  v8f acc = {0.f, 0.f, 0.f, 0.f, 0.f, 0.f, 0.f, 0.f};
  for (int kt = 0; kt < ktiles; ++kt){
    v16bf a = ldfragA(arow, kt * 32);
    v16bf b = ldfragB(brow, kt * 32);
    acc = __builtin_amdgcn_wmma_f32_16x16x32_bf16(false, a, false, b,
                                                  (short)0, acc, false, false);
  }
  return acc;
}

// C/D layout: n = lane&15 (+16*wave), m = r + 8*(lane>>4)
__device__ inline void stc_bias(float* D, int ld, v8f acc, const float* bias){
  int lane = threadIdx.x & 31, wv = threadIdx.x >> 5;
  int ng = wv * 16 + (lane & 15);
  int m0 = (lane >> 4) * 8;
  float b = bias[ng];
#pragma unroll
  for (int r = 0; r < 8; ++r) D[(m0 + r) * ld + ng] = acc[r] + b;
}

__device__ inline void stc_resrelu(float* D, const float* O, int ld, v8f acc,
                                   const float* bias){
  int lane = threadIdx.x & 31, wv = threadIdx.x >> 5;
  int ng = wv * 16 + (lane & 15);
  int m0 = (lane >> 4) * 8;
  float b = bias[ng];
#pragma unroll
  for (int r = 0; r < 8; ++r){
    float x = acc[r] + b;
    x = x > 0.f ? x : 0.f;
    int m = m0 + r;
    D[m * ld + ng] = O[m * ld + ng] + x;
  }
}

// Stage f32 activation [mvalid,128] into bf16 A buffer ld=136, zero-padded.
__device__ inline void stage_abf(unsigned short* sAbf, const float* S,
                                 int lds_, int mvalid){
  for (int i = threadIdx.x; i < 16 * 136; i += 256){
    int m = i / 136, k = i - m * 136;
    float f = (m < mvalid && k < 128) ? S[m * lds_ + k] : 0.f;
    sAbf[i] = f2bf(f);
  }
}

// 4-head attention: q:[n,128](ld=ldq), k/v:[m,128](ld=132) -> o = q + A@v
__device__ inline void attn(float* sS, const float* q, int ldq,
                            const float* k, const float* v, float* o,
                            int n, int m){
  const int t = threadIdx.x;
  const float scale = 0.088388347648318447f;  // 1/sqrt(128)
  int combos = 4 * n * m;
  if (t < combos){
    int h = t / (n * m);
    int rem = t - h * (n * m);
    int i = rem / m, j = rem - (rem / m) * m;
    const float* qp = q + i * ldq + h * 32;
    const float* kp = k + j * 132 + h * 32;
    float s = 0.f;
    for (int d = 0; d < 32; ++d) s += qp[d] * kp[d];
    sS[h * 256 + i * 16 + j] = s * scale;
  }
  __syncthreads();
  int rows = 4 * n;
  if (t < rows){
    int h = t / n, i = t - h * n;
    float* row = sS + h * 256 + i * 16;
    float mx = row[0];
    for (int j = 1; j < m; ++j) mx = fmaxf(mx, row[j]);
    float sum = 0.f;
    for (int j = 0; j < m; ++j){ float e = __expf(row[j] - mx); row[j] = e; sum += e; }
    float inv = 1.f / sum;
    for (int j = 0; j < m; ++j) row[j] *= inv;
  }
  __syncthreads();
  int outs = n * 128;
  for (int idx = t; idx < outs; idx += 256){
    int i = idx >> 7, c = idx & 127, h = c >> 5;
    float acc = q[i * ldq + c];
    const float* arow = sS + h * 256 + i * 16;
    for (int j = 0; j < m; ++j) acc += arow[j] * v[j * 132 + c];
    o[i * 132 + c] = acc;
  }
  __syncthreads();
}

// ------------------------- prep: weights -> bf16 transposed + const q -----
struct PrepArgs {
  const float* srcs[13];
  unsigned short* WT;
  const float* wk0; const float* wv0;
  unsigned short* WkT0; unsigned short* WvT0;
  const float* I1; const float* wq0; const float* bq0;
  const float* I2; const float* wq2; const float* bq2;
  float* q0c; float* q2c;
};

__global__ __launch_bounds__(256) void hnhn_prep(PrepArgs P){
  int b = blockIdx.x, t = threadIdx.x;
  if (b < 13){
    const float* src = P.srcs[b];
    unsigned short* dst = P.WT + b * 16384;
    for (int i = t; i < 16384; i += 256){
      int n = i >> 7, k = i & 127;
      dst[i] = f2bf(src[k * 128 + n]);
    }
  } else if (b == 13 || b == 14){
    const float* src = (b == 13) ? P.wk0 : P.wv0;          // [48,128] f32
    unsigned short* dst = (b == 13) ? P.WkT0 : P.WvT0;     // [128][64] bf16
    for (int i = t; i < 8192; i += 256){
      int n = i >> 6, k = i & 63;
      dst[i] = (k < DW_N) ? f2bf(src[k * 128 + n]) : (unsigned short)0;
    }
  } else {
    for (int i = t; i < 512; i += 256){
      int r = i >> 7, n = i & 127;
      float a0 = P.bq0[n], a2 = P.bq2[n];
      for (int d = 0; d < 128; ++d){
        a0 += P.I1[r * 128 + d] * P.wq0[d * 128 + n];
        a2 += P.I2[r * 128 + d] * P.wq2[d * 128 + n];
      }
      P.q0c[i] = a0; P.q2c[i] = a2;
    }
  }
}

__global__ __launch_bounds__(256) void hnhn_zero(float* p, int n){
  int i = blockIdx.x * blockDim.x + threadIdx.x;
  int stride = gridDim.x * blockDim.x;
  for (; i < n; i += stride) p[i] = 0.f;
}

// ------------------------- main: per-hyperedge pipeline -------------------
struct MainArgs {
  const int* members; const float* vfeat; const float* pw;
  const float* ereg; const float* vreg;
  const float* bk0; const float* bv0; const float* bo0;
  const float* bq1; const float* bk1; const float* bv1; const float* bo1;
  const float* bk2; const float* bv2; const float* bo2;
  const float* bq3; const float* bk3; const float* bv3; const float* bo3;
  const float* decw; const float* decb;
  const unsigned short* WkT0; const unsigned short* WvT0;
  const unsigned short* WT;           // 13 x [128][128] bf16 transposed
  const float* q0c; const float* q2c;
  float* aggr; float* out_e;
};

__global__ __launch_bounds__(256) void hnhn_main(MainArgs A){
  __shared__ alignas(16) float sQ[16 * 132];
  __shared__ alignas(16) float sK[16 * 132];
  __shared__ alignas(16) float sV[16 * 132];
  __shared__ alignas(16) float sO[16 * 132];
  __shared__ alignas(16) unsigned short sAbf[16 * 136];
  __shared__ float sS[1024];
  __shared__ float sQ0C[512];
  __shared__ float sQ2C[512];
  __shared__ float sEF[128];
  __shared__ float sWk[16];
  __shared__ int sMem[16];

  const int e = blockIdx.x;
  const int t = threadIdx.x;

  if (t < 16) sMem[t] = A.members[e * K_N + t];
  __syncthreads();
  for (int i = t; i < 512; i += 256){ sQ0C[i] = A.q0c[i]; sQ2C[i] = A.q2c[i]; }
  // stage pair_weight [16,48] -> bf16, ld=72, K padded to 64
  for (int i = t; i < 16 * 72; i += 256){
    int m = i / 72, k = i - m * 72;
    float f = (k < DW_N) ? A.pw[(e * K_N + m) * DW_N + k] : 0.f;
    sAbf[i] = f2bf(f);
  }
  // gather node features [16,128] -> sQ
  for (int i = t; i < 2048; i += 256){
    int m = i >> 7, c = i & 127;
    sQ[m * 132 + c] = A.vfeat[(size_t)sMem[m] * 128 + c];
  }
  if (t < 16) sWk[t] = A.ereg[e] / A.vreg[sMem[t]];
  __syncthreads();

  // ---- imab0: H = MAB(I1, pair_weight) ----
  { v8f a = mm16(sAbf, 72, A.WkT0, 64, 2); stc_bias(sK, 132, a, A.bk0); }
  { v8f a = mm16(sAbf, 72, A.WvT0, 64, 2); stc_bias(sV, 132, a, A.bv0); }
  __syncthreads();
  attn(sS, sQ0C, 128, sK, sV, sO, 4, 16);
  stage_abf(sAbf, sO, 132, 4);
  __syncthreads();
  { v8f a = mm16(sAbf, 136, A.WT + 0 * 16384, 128, 4); stc_resrelu(sV, sO, 132, a, A.bo0); }
  __syncthreads();                       // H in sV rows 0..3

  // ---- imab1: v1 = MAB(vg, H) ----
  stage_abf(sAbf, sV, 132, 4);           // A = H
  __syncthreads();
  { v8f a = mm16(sAbf, 136, A.WT + 2 * 16384, 128, 4); stc_bias(sK, 132, a, A.bk1); }
  { v8f a = mm16(sAbf, 136, A.WT + 3 * 16384, 128, 4); stc_bias(sV, 132, a, A.bv1); }
  __syncthreads();
  stage_abf(sAbf, sQ, 132, 16);          // A = vg
  __syncthreads();
  { v8f a = mm16(sAbf, 136, A.WT + 1 * 16384, 128, 4); stc_bias(sQ, 132, a, A.bq1); }
  __syncthreads();
  attn(sS, sQ, 132, sK, sV, sO, 16, 4);
  stage_abf(sAbf, sO, 132, 16);
  __syncthreads();
  { v8f a = mm16(sAbf, 136, A.WT + 4 * 16384, 128, 4); stc_resrelu(sQ, sO, 132, a, A.bo1); }
  __syncthreads();                       // v1 in sQ

  // ---- isab0: H2 = MAB(I2, v1) ----
  stage_abf(sAbf, sQ, 132, 16);          // A = v1
  __syncthreads();
  { v8f a = mm16(sAbf, 136, A.WT + 5 * 16384, 128, 4); stc_bias(sK, 132, a, A.bk2); }
  { v8f a = mm16(sAbf, 136, A.WT + 6 * 16384, 128, 4); stc_bias(sV, 132, a, A.bv2); }
  __syncthreads();
  attn(sS, sQ2C, 128, sK, sV, sO, 4, 16);
  stage_abf(sAbf, sO, 132, 4);
  __syncthreads();
  { v8f a = mm16(sAbf, 136, A.WT + 7 * 16384, 128, 4); stc_resrelu(sV, sO, 132, a, A.bo2); }
  __syncthreads();                       // H2 in sV rows 0..3

  // ---- isab1: v2 = MAB(v1, H2) ----
  stage_abf(sAbf, sV, 132, 4);           // A = H2
  __syncthreads();
  { v8f a = mm16(sAbf, 136, A.WT + 9  * 16384, 128, 4); stc_bias(sK, 132, a, A.bk3); }
  { v8f a = mm16(sAbf, 136, A.WT + 10 * 16384, 128, 4); stc_bias(sV, 132, a, A.bv3); }
  __syncthreads();
  stage_abf(sAbf, sQ, 132, 16);          // A = v1
  __syncthreads();
  { v8f a = mm16(sAbf, 136, A.WT + 8 * 16384, 128, 4); stc_bias(sQ, 132, a, A.bq3); }
  __syncthreads();
  attn(sS, sQ, 132, sK, sV, sO, 16, 4);
  stage_abf(sAbf, sO, 132, 16);
  __syncthreads();
  { v8f a = mm16(sAbf, 136, A.WT + 11 * 16384, 128, 4); stc_resrelu(sQ, sO, 132, a, A.bo3); }
  __syncthreads();                       // v2 in sQ

  // ---- mean pool + dec ----
  if (t < 128){
    float s = 0.f;
    for (int m = 0; m < 16; ++m) s += sQ[m * 132 + t];
    sEF[t] = s * (1.f / 16.f);
  }
  __syncthreads();
  if (t < 128){
    float acc = A.decb[t];
    for (int d = 0; d < 128; ++d) acc += sEF[d] * A.decw[d * 128 + t];
    A.out_e[(size_t)e * 128 + t] = acc;
    sS[t] = acc;                         // keep efeat_new for scatter
  }
  __syncthreads();

  // ---- HNHN weighted scatter edge -> node ----
  for (int it = 0; it < 8; ++it){
    int idx = it * 256 + t;
    int k = idx >> 7, c = idx & 127;
    atomicAdd(&A.aggr[(size_t)sMem[k] * 128 + c], sWk[k] * sS[c]);
  }
}

// ------------------------- ev: vfeat_new = relu(aggr @ evW + b) -----------
__global__ __launch_bounds__(256) void hnhn_ev(const float* aggr,
    const unsigned short* evT, const float* evb, float* outv){
  __shared__ alignas(16) unsigned short sAbf[16 * 136];
  int t = threadIdx.x;
  size_t row0 = (size_t)blockIdx.x * 16;
  for (int i = t; i < 16 * 136; i += 256){
    int m = i / 136, k = i - m * 136;
    float f = (k < 128) ? aggr[(row0 + m) * 128 + k] : 0.f;
    sAbf[i] = f2bf(f);
  }
  __syncthreads();
  v8f acc = mm16(sAbf, 136, evT, 128, 4);
  int lane = t & 31, wv = t >> 5;
  int ng = wv * 16 + (lane & 15);
  int m0 = (lane >> 4) * 8;
  float b = evb[ng];
#pragma unroll
  for (int r = 0; r < 8; ++r){
    float v = acc[r] + b;
    outv[(row0 + m0 + r) * 128 + ng] = v > 0.f ? v : 0.f;
  }
}

// --------------------------------------------------------------------------
extern "C" void kernel_launch(void* const* d_in, const int* in_sizes, int n_in,
                              void* d_out, int out_size, void* d_ws, size_t ws_size,
                              hipStream_t stream){
  (void)in_sizes; (void)n_in; (void)out_size; (void)ws_size;
  // d_in order (depth-first over setup_inputs dict):
  // 0 members 1 vfeat 2 efeat 3 pair_weight 4 e_reg_weight 5 v_reg_sum
  // 6 imab_I 7..14 imab0{q.w,q.b,k.w,k.b,v.w,v.b,o.w,o.b} 15..22 imab1{...}
  // 23 isab_I 24..31 isab0{...} 32..39 isab1{...} 40 dec.w 41 dec.b 42 ev.w 43 ev.b
  char* ws = (char*)d_ws;
  float* aggr = (float*)ws;
  size_t off = (size_t)NV_N * 128 * sizeof(float);
  unsigned short* WkT0 = (unsigned short*)(ws + off); off += 128 * 64 * 2;
  unsigned short* WvT0 = (unsigned short*)(ws + off); off += 128 * 64 * 2;
  unsigned short* WT   = (unsigned short*)(ws + off); off += (size_t)13 * 16384 * 2;
  float* q0c = (float*)(ws + off); off += 512 * 4;
  float* q2c = (float*)(ws + off); off += 512 * 4;

  PrepArgs P;
  P.srcs[0]  = (const float*)d_in[13];  // imab0.o.w
  P.srcs[1]  = (const float*)d_in[15];  // imab1.q.w
  P.srcs[2]  = (const float*)d_in[17];  // imab1.k.w
  P.srcs[3]  = (const float*)d_in[19];  // imab1.v.w
  P.srcs[4]  = (const float*)d_in[21];  // imab1.o.w
  P.srcs[5]  = (const float*)d_in[26];  // isab0.k.w
  P.srcs[6]  = (const float*)d_in[28];  // isab0.v.w
  P.srcs[7]  = (const float*)d_in[30];  // isab0.o.w
  P.srcs[8]  = (const float*)d_in[32];  // isab1.q.w
  P.srcs[9]  = (const float*)d_in[34];  // isab1.k.w
  P.srcs[10] = (const float*)d_in[36];  // isab1.v.w
  P.srcs[11] = (const float*)d_in[38];  // isab1.o.w
  P.srcs[12] = (const float*)d_in[42];  // ev.w
  P.WT = WT;
  P.wk0 = (const float*)d_in[9];  P.wv0 = (const float*)d_in[11];
  P.WkT0 = WkT0; P.WvT0 = WvT0;
  P.I1 = (const float*)d_in[6];  P.wq0 = (const float*)d_in[7];  P.bq0 = (const float*)d_in[8];
  P.I2 = (const float*)d_in[23]; P.wq2 = (const float*)d_in[24]; P.bq2 = (const float*)d_in[25];
  P.q0c = q0c; P.q2c = q2c;
  hnhn_prep<<<16, 256, 0, stream>>>(P);

  hnhn_zero<<<4096, 256, 0, stream>>>(aggr, NV_N * 128);

  MainArgs M;
  M.members = (const int*)d_in[0];
  M.vfeat   = (const float*)d_in[1];
  M.pw      = (const float*)d_in[3];
  M.ereg    = (const float*)d_in[4];
  M.vreg    = (const float*)d_in[5];
  M.bk0 = (const float*)d_in[10]; M.bv0 = (const float*)d_in[12]; M.bo0 = (const float*)d_in[14];
  M.bq1 = (const float*)d_in[16]; M.bk1 = (const float*)d_in[18];
  M.bv1 = (const float*)d_in[20]; M.bo1 = (const float*)d_in[22];
  M.bk2 = (const float*)d_in[27]; M.bv2 = (const float*)d_in[29]; M.bo2 = (const float*)d_in[31];
  M.bq3 = (const float*)d_in[33]; M.bk3 = (const float*)d_in[35];
  M.bv3 = (const float*)d_in[37]; M.bo3 = (const float*)d_in[39];
  M.decw = (const float*)d_in[40]; M.decb = (const float*)d_in[41];
  M.WkT0 = WkT0; M.WvT0 = WvT0; M.WT = WT; M.q0c = q0c; M.q2c = q2c;
  M.aggr = aggr;
  M.out_e = (float*)d_out + (size_t)NV_N * 128;
  hnhn_main<<<E_N, 256, 0, stream>>>(M);

  hnhn_ev<<<NV_N / 16, 256, 0, stream>>>(aggr, WT + 12 * 16384,
                                         (const float*)d_in[43], (float*)d_out);
}
